// attention_block_4913442587248
// MI455X (gfx1250) — matrix operand order repaired and checked by reading, not compiled, not run on hardware
//
#include <hip/hip_runtime.h>

#define B_  16
#define C_  512
#define N_  1024
#define G_  8

typedef __attribute__((ext_vector_type(16))) __bf16 v16bf;
typedef __attribute__((ext_vector_type(2)))  __bf16 v2bf;
typedef __attribute__((ext_vector_type(8)))  float  v8f;
typedef __attribute__((ext_vector_type(4)))  unsigned int v4u;
typedef __attribute__((ext_vector_type(8)))  int v8i;
typedef __attribute__((ext_vector_type(4)))  int v4i;

#if defined(__has_builtin)
#if __has_builtin(__builtin_amdgcn_cvt_pk_bf16_f32)
#define HAVE_PK_BF16 1
#endif
#endif

union FragU {
    unsigned int u[8];
    v16bf bf;
};

__device__ __forceinline__ unsigned short f2bf(float f) {
    unsigned int b = __builtin_bit_cast(unsigned int, f);
    b += 0x7FFFu + ((b >> 16) & 1u);           // round-to-nearest-even
    return (unsigned short)(b >> 16);
}

// pack two f32 -> packed bf16 pair (lo | hi<<16)
__device__ __forceinline__ unsigned pack2bf(float lo, float hi) {
#ifdef HAVE_PK_BF16
    v2bf r = __builtin_amdgcn_cvt_pk_bf16_f32(lo, hi);
    return __builtin_bit_cast(unsigned, r);
#else
    return (unsigned)f2bf(lo) | ((unsigned)f2bf(hi) << 16);
#endif
}

__device__ __forceinline__ v8f v8f_zero() {
    v8f z = {0.f, 0.f, 0.f, 0.f, 0.f, 0.f, 0.f, 0.f};
    return z;
}

// ---------------------------------------------------------------------------
// Tensor Data Mover: 2D bf16 tile  global -> LDS  (ISA cdna5 ch.7/8).
// D# group0: count=1 | lds_addr | global_addr[56:0] | type=2
// D# group1: data_size=1 (2B), tensor dims == tile dims (always in-bounds),
//            tile_dim0=tile_w, tile_dim1=tile_h, dim0_stride=row_stride.
// Issued by one wave; completion tracked with TENSORcnt.
// ---------------------------------------------------------------------------
__device__ __forceinline__ void tdm_load_2d_bf16(unsigned lds_byte,
                                                 const void* gsrc,
                                                 unsigned tile_w, unsigned tile_h,
                                                 unsigned row_stride_elems) {
    unsigned long long ga = (unsigned long long)(size_t)gsrc;
    v4u g0;
    g0.x = 1u;                                                 // count=1, user D#
    g0.y = lds_byte;                                           // lds_addr (bytes)
    g0.z = (unsigned)ga;                                       // global_addr[31:0]
    g0.w = (unsigned)((ga >> 32) & 0x01FFFFFFu) | (2u << 30);  // [56:32] | type=2
    v8i g1;
    g1[0] = (int)(1u << 16);                 // data_size = 2 bytes
    g1[1] = (int)(tile_w << 16);             // tensor_dim0[15:0]
    g1[2] = (int)(tile_h << 16);             // tensor_dim0 hi | tensor_dim1[15:0]
    g1[3] = (int)(tile_w << 16);             // tensor_dim1 hi | tile_dim0
    g1[4] = (int)tile_h;                     // tile_dim1 (tile_dim2 = 0)
    g1[5] = (int)row_stride_elems;           // tensor_dim0_stride[31:0] (elems)
    g1[6] = 0;
    g1[7] = 0;
    v4i z4 = {0, 0, 0, 0};
    v8i z8 = {0, 0, 0, 0, 0, 0, 0, 0};
    __builtin_amdgcn_tensor_load_to_lds(g0, g1, z4, z4, z8, 0);
}

__device__ __forceinline__ unsigned lds_off_of(const void* p) {
    return (unsigned)(unsigned long long)(size_t)p;   // low 32 bits = LDS offset
}

// ---------------------------------------------------------------------------
// Kernel 0: one-shot f32 -> bf16 conversion of both weight matrices (paired).
// ---------------------------------------------------------------------------
__global__ __launch_bounds__(256)
void wconv_k(const float* __restrict__ wq, const float* __restrict__ wp,
             unsigned int* __restrict__ wqb, unsigned int* __restrict__ wpb) {
    const int i = blockIdx.x * 256 + threadIdx.x;     // one bf16 pair per thread
    const int NQ2 = (3 * C_ * C_) / 2;
    if (i < NQ2) {
        float2 v = *(const float2*)(wq + 2 * i);
        wqb[i] = pack2bf(v.x, v.y);
    } else {
        float2 v = *(const float2*)(wp + 2 * (i - NQ2));
        wpb[i - NQ2] = pack2bf(v.x, v.y);
    }
}

// ---------------------------------------------------------------------------
// Kernel 1: GroupNorm statistics. One block per (b, group): 64 ch * 1024 px.
// ---------------------------------------------------------------------------
__global__ __launch_bounds__(256)
void gn_stats_k(const float* __restrict__ x, float* __restrict__ mean,
                float* __restrict__ rstd) {
    const int bg = blockIdx.x;                           // 0..127
    const float* p = x + (size_t)bg * (C_ / G_) * N_;    // contiguous 65536 floats
    const int tid = threadIdx.x;
    float s = 0.f, ss = 0.f;
    const float4* p4 = (const float4*)p;
    for (int i = tid; i < (C_ / G_) * N_ / 4; i += 256) {
        float4 v = p4[i];
        s  += v.x + v.y + v.z + v.w;
        ss += v.x * v.x + v.y * v.y + v.z * v.z + v.w * v.w;
    }
#pragma unroll
    for (int m = 16; m >= 1; m >>= 1) {
        s  += __shfl_xor(s,  m, 32);
        ss += __shfl_xor(ss, m, 32);
    }
    __shared__ float sh1[8], sh2[8];
    const int wave = tid >> 5;
    if ((tid & 31) == 0) { sh1[wave] = s; sh2[wave] = ss; }
    __syncthreads();
    if (tid == 0) {
        float a = 0.f, b = 0.f;
#pragma unroll
        for (int i = 0; i < 8; ++i) { a += sh1[i]; b += sh2[i]; }
        const float inv = 1.0f / ((float)((C_ / G_) * N_));
        float m  = a * inv;
        float var = b * inv - m * m;
        mean[bg] = m;
        rstd[bg] = rsqrtf(var + 1e-5f);
    }
}

// ---------------------------------------------------------------------------
// Kernel 2: apply GroupNorm, write h transposed as bf16  ht[b][n][c].
// ---------------------------------------------------------------------------
__global__ __launch_bounds__(256)
void gn_apply_k(const float* __restrict__ x, const float* __restrict__ gamma,
                const float* __restrict__ beta, const float* __restrict__ mean,
                const float* __restrict__ rstd, unsigned short* __restrict__ ht) {
    const int idx = blockIdx.x * 256 + threadIdx.x;      // 4 elements per thread
    const int e0  = idx * 4;
    const int b   = e0 >> 19;                            // / (C_*N_)
    const int rem = e0 & (C_ * N_ - 1);
    const int c   = rem >> 10;
    const int n0  = rem & (N_ - 1);
    const int bg  = b * G_ + (c >> 6);
    const float m = mean[bg], rs = rstd[bg];
    const float ga = gamma[c], be = beta[c];
    float4 v = *(const float4*)(x + e0);
    unsigned short* hp = ht + ((size_t)b * N_ + n0) * C_ + c;
    hp[0 * C_] = f2bf((v.x - m) * rs * ga + be);
    hp[1 * C_] = f2bf((v.y - m) * rs * ga + be);
    hp[2 * C_] = f2bf((v.z - m) * rs * ga + be);
    hp[3 * C_] = f2bf((v.w - m) * rs * ga + be);
}

// ---------------------------------------------------------------------------
// Kernel 3: QKV GEMM (bf16 weights).  TDM double-buffered LDS staging.
// block tile 64(o) x 64(n), 4 waves, K-steps of 32.
// q,k stored transposed [b][n][c]; v stored [b][c][n].
// ---------------------------------------------------------------------------
__global__ __launch_bounds__(128)
void qkv_gemm_k(const unsigned short* __restrict__ wq, const float* __restrict__ bias,
                const unsigned short* __restrict__ ht,
                unsigned short* __restrict__ qt, unsigned short* __restrict__ kt,
                unsigned short* __restrict__ vv) {
    const int b  = blockIdx.z;
    const int ob = blockIdx.y * 64;
    const int nb = blockIdx.x * 64;
    const int tid = threadIdx.x, wave = tid >> 5, lane = tid & 31;
    const int mrow = lane & 15;
    const int rowh = (lane >> 4) * 8;
    const int khB  = (lane >> 4) * 16;

    __shared__ unsigned short as[2][64][32];   // A tile (o x c)
    __shared__ unsigned short hs[2][64][32];   // B tile (n x c)

    const unsigned short* htb  = ht + (size_t)b * N_ * C_;
    const unsigned short* arow = wq + (size_t)ob * C_;
    const unsigned short* hrow = htb + (size_t)nb * C_;

    float bi[8];
#pragma unroll
    for (int s = 0; s < 8; ++s) bi[s] = bias[ob + 16 * wave + rowh + s];

    v8f acc[4];
#pragma unroll
    for (int t = 0; t < 4; ++t) acc[t] = v8f_zero();

    if (wave == 0) {
        tdm_load_2d_bf16(lds_off_of(&as[0][0][0]), arow, 32, 64, C_);
        tdm_load_2d_bf16(lds_off_of(&hs[0][0][0]), hrow, 32, 64, C_);
    }

    for (int kc = 0; kc < C_; kc += 32) {
        const int buf = (kc >> 5) & 1;
        if (wave == 0) {
            if (kc + 32 < C_) {
                tdm_load_2d_bf16(lds_off_of(&as[buf ^ 1][0][0]), arow + kc + 32, 32, 64, C_);
                tdm_load_2d_bf16(lds_off_of(&hs[buf ^ 1][0][0]), hrow + kc + 32, 32, 64, C_);
                __builtin_amdgcn_s_wait_tensorcnt(2);   // current tile done
            } else {
                __builtin_amdgcn_s_wait_tensorcnt(0);
            }
        }
        __syncthreads();

        FragU A;
#pragma unroll
        for (int s = 0; s < 8; ++s) {
            int k0 = khB + 2 * s;
            A.u[s] = *(const unsigned int*)&as[buf][16 * wave + mrow][k0];
        }
#pragma unroll
        for (int t = 0; t < 4; ++t) {
            FragU Bf;
#pragma unroll
            for (int s = 0; s < 8; ++s)
                Bf.u[s] = *(const unsigned int*)&hs[buf][16 * t + mrow][khB + 2 * s];
            acc[t] = __builtin_amdgcn_wmma_f32_16x16x32_bf16(
                false, A.bf, false, Bf.bf, (short)0, acc[t], false, false);
        }
        __syncthreads();
    }

    const int o0 = ob + 16 * wave + rowh;
#pragma unroll
    for (int t = 0; t < 4; ++t) {
        const int n = nb + 16 * t + mrow;
        if (ob < 1024) {                     // q / k: transposed layout, vector store
            unsigned short* dst = (ob < 512)
                ? qt + ((size_t)b * N_ + n) * C_ + o0
                : kt + ((size_t)b * N_ + n) * C_ + (o0 - 512);
            uint4 u;
            u.x = pack2bf(acc[t][0] + bi[0], acc[t][1] + bi[1]);
            u.y = pack2bf(acc[t][2] + bi[2], acc[t][3] + bi[3]);
            u.z = pack2bf(acc[t][4] + bi[4], acc[t][5] + bi[5]);
            u.w = pack2bf(acc[t][6] + bi[6], acc[t][7] + bi[7]);
            *(uint4*)dst = u;
        } else {                             // v: [c][n] layout, scattered b16 stores
#pragma unroll
            for (int s = 0; s < 8; ++s)
                vv[((size_t)b * C_ + (o0 - 1024 + s)) * N_ + n] = f2bf(acc[t][s] + bi[s]);
        }
    }
}

// ---------------------------------------------------------------------------
// Kernel 4: single-pass flash attention, TDM staging.
// grid = (c-half, i-strip, b); block = 4 waves; wave owns 16 rows x 256 cols of O.
// ---------------------------------------------------------------------------
__global__ __launch_bounds__(128)
void attn_flash_k(const unsigned short* __restrict__ qt,
                  const unsigned short* __restrict__ kt,
                  const unsigned short* __restrict__ vv,
                  unsigned short* __restrict__ oa /* [b][c][n] */) {
    const int chalf = blockIdx.x * 256;
    const int ib    = blockIdx.y * 64;
    const int b     = blockIdx.z;
    const int tid = threadIdx.x, wave = tid >> 5, lane = tid & 31;
    const int mrow = lane & 15;
    const int rowh = (lane >> 4) * 8;
    const int khB  = (lane >> 4) * 16;

    __shared__ unsigned short qs[64][64];     // q chunk  (i x c64)
    __shared__ unsigned short ks[64][64];     // k chunk  (j x c64)
    __shared__ unsigned short vs[256 * 64];   // v tile   (c256 x j64)
    __shared__ unsigned short ps[64 * 64];    // softmax weights (i x j)

    const unsigned short* qtb = qt + (size_t)b * N_ * C_;
    const unsigned short* ktb = kt + (size_t)b * N_ * C_;
    const unsigned short* vvb = vv + (size_t)b * C_ * N_;

    v8f oacc[16];
#pragma unroll
    for (int t = 0; t < 16; ++t) oacc[t] = v8f_zero();
    float mr[8], lr[8];
#pragma unroll
    for (int s = 0; s < 8; ++s) { mr[s] = -1e30f; lr[s] = 0.0f; }

    const float scale = 0.044194173824159216f;   // 1/sqrt(512)

    for (int jb = 0; jb < N_; jb += 64) {
        __syncthreads();                          // prior readers of vs/ps done
        if (wave == 0)                            // V half tile rides ahead of q/k
            tdm_load_2d_bf16(lds_off_of(vs), vvb + (size_t)chalf * N_ + jb,
                             64, 256, N_);

        // ---- S = (q^T k) over full C, chunked by 64 ----
        v8f sacc[4];
#pragma unroll
        for (int t = 0; t < 4; ++t) sacc[t] = v8f_zero();

        for (int kc = 0; kc < C_; kc += 64) {
            __syncthreads();                      // prior chunk gathers done
            if (wave == 0) {
                tdm_load_2d_bf16(lds_off_of(qs), qtb + (size_t)ib * C_ + kc, 64, 64, C_);
                tdm_load_2d_bf16(lds_off_of(ks), ktb + (size_t)jb * C_ + kc, 64, 64, C_);
                __builtin_amdgcn_s_wait_tensorcnt(0);
            }
            __syncthreads();
#pragma unroll
            for (int k2 = 0; k2 < 64; k2 += 32) {
                FragU A;
#pragma unroll
                for (int s = 0; s < 8; ++s) {
                    int k0 = khB + 2 * s;
                    A.u[s] = *(const unsigned int*)&qs[16 * wave + mrow][k2 + k0];
                }
#pragma unroll
                for (int t = 0; t < 4; ++t) {
                    FragU Bf;
#pragma unroll
                    for (int s = 0; s < 8; ++s)
                        Bf.u[s] = *(const unsigned int*)&ks[16 * t + mrow][k2 + khB + 2 * s];
                    sacc[t] = __builtin_amdgcn_wmma_f32_16x16x32_bf16(
                        false, A.bf, false, Bf.bf, (short)0, sacc[t], false, false);
                }
            }
        }

        // ---- online softmax (rows live in VGPR slots, cols in lanes) ----
        const int iloc = 16 * wave + rowh;
#pragma unroll
        for (int s = 0; s < 8; ++s) {
            float mx = fmaxf(fmaxf(sacc[0][s], sacc[1][s]),
                             fmaxf(sacc[2][s], sacc[3][s])) * scale;
#pragma unroll
            for (int msk = 1; msk <= 8; msk <<= 1) mx = fmaxf(mx, __shfl_xor(mx, msk, 32));
            float nm   = fmaxf(mr[s], mx);
            float corr = __expf(mr[s] - nm);
            float psum = 0.f;
#pragma unroll
            for (int t = 0; t < 4; ++t) {
                float p = __expf(sacc[t][s] * scale - nm);
                psum += p;
                ps[(iloc + s) * 64 + 16 * t + mrow] = f2bf(p);
            }
#pragma unroll
            for (int msk = 1; msk <= 8; msk <<= 1) psum += __shfl_xor(psum, msk, 32);
            lr[s] = lr[s] * corr + psum;
            mr[s] = nm;
#pragma unroll
            for (int ct = 0; ct < 16; ++ct) oacc[ct][s] *= corr;
        }
        __syncthreads();

        // ---- O += P @ V^T  (K = j, 2 chunks of 32; N = c, 16 tiles) ----
#pragma unroll
        for (int kj = 0; kj < 64; kj += 32) {
            FragU A;
#pragma unroll
            for (int s = 0; s < 8; ++s) {
                int k0 = khB + 2 * s;
                A.u[s] = *(const unsigned int*)&ps[(16 * wave + mrow) * 64 + kj + k0];
            }
#pragma unroll
            for (int ct = 0; ct < 16; ++ct) {
                FragU Bf;
#pragma unroll
                for (int s = 0; s < 8; ++s)
                    Bf.u[s] = *(const unsigned int*)&vs[(16 * ct + mrow) * 64 + kj + khB + 2 * s];
                oacc[ct] = __builtin_amdgcn_wmma_f32_16x16x32_bf16(
                    false, A.bf, false, Bf.bf, (short)0, oacc[ct], false, false);
            }
        }
    }

    // epilogue: O *= 1/l (v_rcp), store bf16 to oa[b][c][i]
    float il[8];
#pragma unroll
    for (int s = 0; s < 8; ++s) il[s] = __builtin_amdgcn_rcpf(lr[s]);
#pragma unroll
    for (int ct = 0; ct < 16; ++ct) {
        const int c = chalf + 16 * ct + mrow;
        uint4 u;
        u.x = pack2bf(oacc[ct][0] * il[0], oacc[ct][1] * il[1]);
        u.y = pack2bf(oacc[ct][2] * il[2], oacc[ct][3] * il[3]);
        u.z = pack2bf(oacc[ct][4] * il[4], oacc[ct][5] * il[5]);
        u.w = pack2bf(oacc[ct][6] * il[6], oacc[ct][7] * il[7]);
        *(uint4*)(oa + ((size_t)b * C_ + c) * N_ + ib + 16 * wave + rowh) = u;
    }
}

// ---------------------------------------------------------------------------
// Kernel 5: projection GEMM + residual (bf16 weights, TDM double-buffered).
// ---------------------------------------------------------------------------
__global__ __launch_bounds__(128)
void proj_gemm_k(const unsigned short* __restrict__ wp, const float* __restrict__ bias,
                 const unsigned short* __restrict__ oa,
                 const float* __restrict__ x, float* __restrict__ out) {
    const int b  = blockIdx.z;
    const int ob = blockIdx.y * 64;
    const int nb = blockIdx.x * 64;
    const int tid = threadIdx.x, wave = tid >> 5, lane = tid & 31;
    const int mrow = lane & 15;
    const int rowh = (lane >> 4) * 8;
    const int khB  = (lane >> 4) * 16;

    __shared__ unsigned short as[2][64][32];   // w_proj tile (o x c)
    __shared__ unsigned short os[2][32][64];   // oa tile (c x n)

    const unsigned short* oab  = oa + (size_t)b * C_ * N_;
    const unsigned short* arow = wp + (size_t)ob * C_;

    float bi[8];
#pragma unroll
    for (int s = 0; s < 8; ++s) bi[s] = bias[ob + 16 * wave + rowh + s];

    v8f acc[4];
#pragma unroll
    for (int t = 0; t < 4; ++t) acc[t] = v8f_zero();

    if (wave == 0) {
        tdm_load_2d_bf16(lds_off_of(&as[0][0][0]), arow, 32, 64, C_);
        tdm_load_2d_bf16(lds_off_of(&os[0][0][0]), oab + nb, 64, 32, N_);
    }

    for (int kc = 0; kc < C_; kc += 32) {
        const int buf = (kc >> 5) & 1;
        if (wave == 0) {
            if (kc + 32 < C_) {
                tdm_load_2d_bf16(lds_off_of(&as[buf ^ 1][0][0]), arow + kc + 32, 32, 64, C_);
                tdm_load_2d_bf16(lds_off_of(&os[buf ^ 1][0][0]),
                                 oab + (size_t)(kc + 32) * N_ + nb, 64, 32, N_);
                __builtin_amdgcn_s_wait_tensorcnt(2);
            } else {
                __builtin_amdgcn_s_wait_tensorcnt(0);
            }
        }
        __syncthreads();

        FragU A;
#pragma unroll
        for (int s = 0; s < 8; ++s) {
            int k0 = khB + 2 * s;
            A.u[s] = *(const unsigned int*)&as[buf][16 * wave + mrow][k0];
        }
#pragma unroll
        for (int t = 0; t < 4; ++t) {
            FragU Bf;
            const int n = 16 * t + mrow;
#pragma unroll
            for (int s = 0; s < 8; ++s) {
                unsigned int lo = os[buf][khB + 2 * s][n];
                unsigned int hi = os[buf][khB + 2 * s + 1][n];
                Bf.u[s] = lo | (hi << 16);
            }
            acc[t] = __builtin_amdgcn_wmma_f32_16x16x32_bf16(
                false, A.bf, false, Bf.bf, (short)0, acc[t], false, false);
        }
        __syncthreads();
    }

#pragma unroll
    for (int t = 0; t < 4; ++t) {
        const int n = nb + 16 * t + mrow;
#pragma unroll
        for (int s = 0; s < 8; ++s) {
            const int o = ob + 16 * wave + rowh + s;
            const size_t off = ((size_t)b * C_ + o) * N_ + n;
            out[off] = x[off] + acc[t][s] + bi[s];
        }
    }
}

// ---------------------------------------------------------------------------
extern "C" void kernel_launch(void* const* d_in, const int* in_sizes, int n_in,
                              void* d_out, int out_size, void* d_ws, size_t ws_size,
                              hipStream_t stream) {
    const float* x      = (const float*)d_in[0];
    const float* gamma  = (const float*)d_in[1];
    const float* beta   = (const float*)d_in[2];
    const float* w_qkv  = (const float*)d_in[3];
    const float* b_qkv  = (const float*)d_in[4];
    const float* w_proj = (const float*)d_in[5];
    const float* b_proj = (const float*)d_in[6];
    float* out = (float*)d_out;

    char* ws = (char*)d_ws;
    float* stats = (float*)ws;                          // mean[128], rstd[128]
    unsigned short* wqb = (unsigned short*)(ws + 4096); // bf16 w_qkv  [1536][512]
    unsigned short* wpb = wqb + (size_t)3 * C_ * C_;    // bf16 w_proj [512][512]
    const size_t TENSOR = (size_t)B_ * N_ * C_;         // 8M bf16 elements (16 MB)
    unsigned short* ht = wpb + (size_t)C_ * C_;
    unsigned short* qt = ht + TENSOR;
    unsigned short* kt = qt + TENSOR;
    unsigned short* vv = kt + TENSOR;
    unsigned short* oa = ht;                            // reuse ht region after QKV

    wconv_k<<<(4 * C_ * C_) / (2 * 256), 256, 0, stream>>>(
        w_qkv, w_proj, (unsigned int*)wqb, (unsigned int*)wpb);
    gn_stats_k<<<B_ * G_, 256, 0, stream>>>(x, stats, stats + 128);
    gn_apply_k<<<(B_ * C_ * N_) / (256 * 4), 256, 0, stream>>>(
        x, gamma, beta, stats, stats + 128, ht);
    qkv_gemm_k<<<dim3(N_ / 64, (3 * C_) / 64, B_), 128, 0, stream>>>(
        wqb, b_qkv, ht, qt, kt, vv);
    attn_flash_k<<<dim3(C_ / 256, N_ / 64, B_), 128, 0, stream>>>(qt, kt, vv, oa);
    proj_gemm_k<<<dim3(N_ / 64, C_ / 64, B_), 128, 0, stream>>>(
        wpb, b_proj, oa, x, out);
}
